// Head_84043920048318
// MI455X (gfx1250) — matrix operand order repaired and checked by reading, not compiled, not run on hardware
//
#include <hip/hip_runtime.h>
#include <hip/hip_bf16.h>

// ---------------------------------------------------------------------------
// B=32, S=8, T=512, C=128, HS=128
// out[b,s,t,h] = softmax_u( (q_b t . k_b u) * C^-0.5 * adjacent[b,s,t,u] ) @ v
// HBM-bound on adjacent (268 MB @ 23.3 TB/s ~ 15 us); matmuls on
// v_wmma_f32_16x16x32_f16; V staged to LDS via global_load_async_to_lds_b128.
// ---------------------------------------------------------------------------

typedef __attribute__((ext_vector_type(16))) _Float16 v16h;
typedef __attribute__((ext_vector_type(8)))  float    v8f;

#define BB  32
#define SS  8
#define TT  512
#define CC  128
#define HSZ 128
#define VT_STRIDE 520              // padded LDS stride (halves); 1040B = 65*16B

// A-matrix (16xK f16) per-lane K offset within a 32-wide K chunk.
__device__ __forceinline__ int kOff(int i, int half) {
    return half * 16 + i;
}

__device__ __forceinline__ v8f wmma_f16(v16h a, v16h b, v8f c) {
    return __builtin_amdgcn_wmma_f32_16x16x32_f16(false, a, false, b,
                                                  (short)0, c, false, false);
}

// ---------------------------------------------------------------------------
// Kernel 1: q/k/v = f16( x @ W{q,k,v} ); scale C^-0.5 folded into q.
// q,k stored row-major [B][T][HS]; v stored TRANSPOSED [B][HS][T] so the
// attention kernel can stage it to LDS with contiguous 16B async copies.
// One wave per 16x16 output tile, K=128 contraction = 4 WMMA.
// ---------------------------------------------------------------------------
__global__ void qkv_proj_kernel(const float* __restrict__ x,
                                const float* __restrict__ Wq,
                                const float* __restrict__ Wk,
                                const float* __restrict__ Wv,
                                _Float16* __restrict__ qo,
                                _Float16* __restrict__ ko,
                                _Float16* __restrict__ vo) {
    const int lane = threadIdx.x & 31;
    const int wave = threadIdx.x >> 5;
    const int half = lane >> 4;
    const int ln   = lane & 15;

    const int job = blockIdx.x * 8 + wave;      // 0 .. 24575
    const int mat = job / 8192;                 // 0=q 1=k 2=v
    const int rem = job % 8192;
    const int rowTile = rem / 8;                // 1024 row tiles of 16
    const int colTile = rem % 8;                // 8 col tiles of 16
    const int r0 = rowTile * 16;                // row in flattened [B*T]
    const int c0 = colTile * 16;

    const float* W = (mat == 0) ? Wq : ((mat == 1) ? Wk : Wv);

    // A: 16 rows of x (f32 -> f16), 4 chunks of K=32
    v16h a[4];
    const float* xr = x + (size_t)(r0 + ln) * CC;
    #pragma unroll
    for (int kb = 0; kb < 4; ++kb)
        #pragma unroll
        for (int i = 0; i < 16; ++i)
            a[kb][i] = (_Float16)xr[kb * 32 + kOff(i, half)];

    v8f acc = {};
    #pragma unroll
    for (int kb = 0; kb < 4; ++kb) {
        // B: W[kb*32 + kk][c0 + n], kk = half*16 + i, n = ln
        v16h bm;
        const float* wp = W + (size_t)(kb * 32 + half * 16) * HSZ + c0 + ln;
        #pragma unroll
        for (int i = 0; i < 16; ++i)
            bm[i] = (_Float16)wp[(size_t)i * HSZ];
        acc = wmma_f16(a[kb], bm, acc);
    }

    if (mat == 2) {
        // V transposed: vo[b][h][t]; lane's C-tile column (N=ln) is a run of
        // 8 consecutive t's -> one contiguous 16B store per lane.
        const int b  = r0 / TT;
        const int t0 = (r0 % TT) + half * 8;
        _Float16* vt = vo + (size_t)b * HSZ * TT + (size_t)(c0 + ln) * TT + t0;
        #pragma unroll
        for (int r = 0; r < 8; ++r)
            vt[r] = (_Float16)acc[r];
    } else {
        _Float16* outp = (mat == 0) ? qo : ko;
        const float scl = (mat == 0) ? 0.08838834764831845f : 1.0f;  // 128^-0.5
        #pragma unroll
        for (int r = 0; r < 8; ++r)
            outp[(size_t)(r0 + r + half * 8) * HSZ + c0 + ln] =
                (_Float16)(acc[r] * scl);
    }
}

// ---------------------------------------------------------------------------
// Kernel 2: fused scores * adjacent -> exp -> @ v, streaming adjacent once.
// Block: 256 thr (8 wave32), 128 query rows, one (b,s). V_b^T staged into
// LDS with per-lane async 16B copies (ASYNCcnt); per-wave LDS scratch for the
// P C-layout -> A-layout transpose (DScnt). grid = B*S*(T/128) = 1024.
// ---------------------------------------------------------------------------
__global__ void attn_kernel(const float* __restrict__ adjacent,
                            const _Float16* __restrict__ q,
                            const _Float16* __restrict__ k,
                            const _Float16* __restrict__ v,
                            float* __restrict__ out) {
    extern __shared__ char smem_raw[];
    _Float16* vT   = (_Float16*)smem_raw;            // [HS][VT_STRIDE]
    _Float16* pscr = vT + HSZ * VT_STRIDE;           // 8 waves * 16*32 halves

    const int tid  = threadIdx.x;
    const int lane = tid & 31;
    const int wave = tid >> 5;
    const int half = lane >> 4;
    const int ln   = lane & 15;

    const int bid = blockIdx.x;
    const int rt  = bid & 3;                 // row tile (128 rows)
    const int s   = (bid >> 2) & 7;
    const int b   = bid >> 5;
    const int r0w = rt * 128 + wave * 16;    // this wave's first query row

    // ---- stage V_b^T into LDS: contiguous 16B global -> 16B LDS async copies
    {
        const _Float16* vbT = v + (size_t)b * HSZ * TT;   // [HS][T]
        for (int cidx = tid; cidx < (HSZ * TT) / 8; cidx += 256) {
            const int h   = cidx >> 6;            // 64 16B-chunks per h-row
            const int t8  = (cidx & 63) << 3;     // t offset (halves)
            const _Float16* g = vbT + (size_t)h * TT + t8;
            const unsigned ldsb =
                (unsigned)((h * VT_STRIDE + t8) * sizeof(_Float16));
            asm volatile("global_load_async_to_lds_b128 %0, %1, off"
                         :: "v"(ldsb), "v"(g) : "memory");
        }
        asm volatile("s_wait_asynccnt 0" ::: "memory");
    }

    // ---- preload this wave's 16 q rows in A layout (K = 128 -> 4 chunks)
    const _Float16* qb = q + (size_t)(b * TT + r0w) * HSZ;
    v16h aq[4];
    #pragma unroll
    for (int kb = 0; kb < 4; ++kb)
        #pragma unroll
        for (int i = 0; i < 16; ++i)
            aq[kb][i] = qb[(size_t)ln * HSZ + kb * 32 + kOff(i, half)];

    __syncthreads();   // vT visible to all waves

    const _Float16* kbase = k + (size_t)b * TT * HSZ;
    const float* adjw = adjacent + (((size_t)(b * SS + s) * TT) + r0w) * TT;

    v8f acc[8];
    #pragma unroll
    for (int c = 0; c < 8; ++c) acc[c] = (v8f){};
    float rs[8] = {0, 0, 0, 0, 0, 0, 0, 0};

    _Float16* ps = pscr + wave * (16 * 32);  // [m][kk] row-major, kk in [0,32)

    for (int kt = 0; kt < TT / 32; ++kt) {
        // prefetch next adjacency tile (global_prefetch_b8)
        if (kt + 1 < TT / 32)
            __builtin_prefetch(&adjw[(size_t)(half * 8) * TT + (kt + 1) * 32 + ln], 0, 0);

        // ---- scores: two 16x16 tiles (keys kt*32 .. +31), K = 128
        v8f sc[2];
        #pragma unroll
        for (int nt = 0; nt < 2; ++nt) {
            sc[nt] = (v8f){};
            const int key0 = kt * 32 + nt * 16;
            #pragma unroll
            for (int kb = 0; kb < 4; ++kb) {
                // B[kk][n] = k[key0+n][kb*32+kk]: contiguous 16 halves/lane
                v16h bk;
                const _Float16* kp =
                    kbase + (size_t)(key0 + ln) * HSZ + kb * 32 + half * 16;
                #pragma unroll
                for (int i = 0; i < 16; ++i) bk[i] = kp[i];
                sc[nt] = wmma_f16(aq[kb], bk, sc[nt]);
            }
        }

        // ---- wei = score*adj, p = exp(wei); row-sum partials; stash P in LDS
        #pragma unroll
        for (int nt = 0; nt < 2; ++nt) {
            const int key0 = kt * 32 + nt * 16;
            #pragma unroll
            for (int r = 0; r < 8; ++r) {
                const int m = r + half * 8;
                const float av = adjw[(size_t)m * TT + key0 + ln];
                const float p  = __expf(sc[nt][r] * av);
                rs[r] += p;
                ps[m * 32 + nt * 16 + ln] = (_Float16)p;
            }
        }
        // same-wave LDS RAW across lanes: drain DS stores before re-reading
        asm volatile("s_wait_dscnt 0" ::: "memory");

        // ---- reload P in A layout (16x32 f16)
        v16h pa;
        #pragma unroll
        for (int i = 0; i < 16; ++i) pa[i] = ps[ln * 32 + kOff(i, half)];

        // ---- acc[c] += P @ V tile; B[kk][n] = vT contiguous 16 halves
        #pragma unroll
        for (int c = 0; c < 8; ++c) {
            v16h bv;
            const _Float16* vp =
                vT + (c * 16 + ln) * VT_STRIDE + kt * 32 + half * 16;
            #pragma unroll
            for (int i = 0; i < 16; ++i) bv[i] = vp[i];
            acc[c] = wmma_f16(pa, bv, acc[c]);
        }
    }

    // ---- softmax denominator: reduce partial sums across the 16 lanes/half
    #pragma unroll
    for (int r = 0; r < 8; ++r) {
        float t = rs[r];
        t += __shfl_xor(t, 1);
        t += __shfl_xor(t, 2);
        t += __shfl_xor(t, 4);
        t += __shfl_xor(t, 8);
        rs[r] = t;
    }

    // ---- write out[b][s][r0w + m][c*16 + ln]
    float* ow = out + (((size_t)(b * SS + s) * TT) + r0w) * HSZ;
    #pragma unroll
    for (int c = 0; c < 8; ++c) {
        #pragma unroll
        for (int r = 0; r < 8; ++r)
            ow[(size_t)(r + half * 8) * HSZ + c * 16 + ln] = acc[c][r] / rs[r];
    }
}

// ---------------------------------------------------------------------------
extern "C" void kernel_launch(void* const* d_in, const int* in_sizes, int n_in,
                              void* d_out, int out_size, void* d_ws, size_t ws_size,
                              hipStream_t stream) {
    const float* x   = (const float*)d_in[0];   // [B,T,C]
    const float* adj = (const float*)d_in[1];   // [B,S,T,T]
    const float* Wq  = (const float*)d_in[2];   // [C,HS]
    const float* Wk  = (const float*)d_in[3];
    const float* Wv  = (const float*)d_in[4];
    float* out = (float*)d_out;                  // [B,S,T,HS]

    const size_t elems = (size_t)BB * TT * HSZ;  // 2,097,152 per matrix
    _Float16* qs = (_Float16*)d_ws;
    _Float16* ks = qs + elems;
    _Float16* vs = ks + elems;                   // stored [B][HS][T]

    qkv_proj_kernel<<<3072, 256, 0, stream>>>(x, Wq, Wk, Wv, qs, ks, vs);

    const size_t smem = (size_t)(HSZ * VT_STRIDE + 8 * 16 * 32) * sizeof(_Float16);
    attn_kernel<<<BB * SS * (TT / 128), 256, smem, stream>>>(adj, qs, ks, vs, out);
}
